// Transformer_81810537054523
// MI455X (gfx1250) — matrix operand order repaired and checked by reading, not compiled, not run on hardware
//
#include <hip/hip_runtime.h>
#include <hip/hip_bf16.h>

// ---------------- model dims ----------------
#define B_   4
#define S_   1024
#define D_   512
#define H_   8
#define DH_  64
#define FF_  2048
#define L_   2
#define MS_  (B_ * S_)          // 4096 rows
#define VOC_ 32000

typedef __attribute__((ext_vector_type(16))) __bf16 v16bf;
typedef __attribute__((ext_vector_type(8)))  float  v8f;
typedef unsigned short u16_t;

// =====================================================================
// Batched GEMM:  C[z] = alpha * A[z] x Bop[z] (+bias) (+relu)
//   A: [M,K] bf16, row stride lda, batch stride sA
//   opB==0:  B given as W[N,K] row-major (C = A @ W^T)
//   opB==1:  B given as V[K,N] row-major (C = A @ V)
//   C: [M,N] f32, row stride ldc, batch stride sC
// Block: 256 thr (8 waves, 4(m) x 2(n)). Tile 128(M) x BN(N), K-step 32.
//   BN = 32*NSUB.  Wave tile: 32 x 16*NSUB  ->  2 x NSUB WMMA accumulators.
// Requires: M%128==0, N%BN==0, K%32==0, strides %8==0 (true for all calls).
// =====================================================================
template <int NSUB>   // 2 -> BN=64, 4 -> BN=128
__global__ __launch_bounds__(256) void k_gemm_bf16(
    const u16_t* __restrict__ A, long long lda, long long sA,
    const u16_t* __restrict__ Bm, long long ldb, long long sB,
    float* __restrict__ C, long long ldc, long long sC,
    int M, int N, int K,
    const float* __restrict__ bias, float alpha, int relu, int opB)
{
    constexpr int BN = 32 * NSUB;
    __shared__ u16_t As[128][40];    // 32 used + 8 pad
    __shared__ u16_t Bs[BN][40];     // stored as [n][k]

    const int tid  = threadIdx.x;
    const int lane = tid & 31;
    const int w    = tid >> 5;
    const int waveM = (w & 3) * 32;
    const int waveN = (w >> 2) * (16 * NSUB);
    const int m0 = blockIdx.y * 128;
    const int n0 = blockIdx.x * BN;
    const long long z = blockIdx.z;

    const u16_t* Ab = A  + z * sA;
    const u16_t* Bb = Bm + z * sB;
    float*       Cb = C  + z * sC;

    v8f acc[2][NSUB];
#pragma unroll
    for (int i = 0; i < 2; ++i)
#pragma unroll
        for (int j = 0; j < NSUB; ++j)
#pragma unroll
            for (int t = 0; t < 8; ++t) acc[i][j][t] = 0.0f;

    for (int kk = 0; kk < K; kk += 32) {
        // ---- stage A tile: 128 rows x 32 k (2 x 16B per thread) ----
#pragma unroll
        for (int i = 0; i < 2; ++i) {
            int c   = tid * 2 + i;
            int row = c >> 2;
            int kc  = (c & 3) * 8;
            uint4 v = *(const uint4*)(Ab + (long long)(m0 + row) * lda + kk + kc);
            *(uint4*)&As[row][kc] = v;
        }
        // ---- stage B tile: BN n x 32 k ----
        if (opB == 0) {
#pragma unroll
            for (int it = 0; it < NSUB / 2; ++it) {
                int c  = it * 256 + tid;
                int n  = c >> 2;
                int kc = (c & 3) * 8;
                uint4 v = *(const uint4*)(Bb + (long long)(n0 + n) * ldb + kk + kc);
                *(uint4*)&Bs[n][kc] = v;
            }
        } else {
            constexpr int NCH = BN / 8;          // n-chunks of 8
#pragma unroll
            for (int it = 0; it < NSUB / 2; ++it) {
                int c  = it * 256 + tid;
                int k  = c / NCH;
                int nc = (c % NCH) * 8;
                uint4 v = *(const uint4*)(Bb + (long long)(kk + k) * ldb + n0 + nc);
                const u16_t* pv = (const u16_t*)&v;
#pragma unroll
                for (int i = 0; i < 8; ++i) Bs[nc + i][k] = pv[i];   // transpose into LDS
            }
        }
        __syncthreads();

        if (kk + 32 < K) {  // pull next A tile toward cache (global_prefetch_b8)
            __builtin_prefetch(Ab + (long long)(m0 + (tid >> 1)) * lda + kk + 32, 0, 3);
        }

        // ---- fragments (ISA 16-bit layouts) ----
        v16bf aF[2], bF[NSUB];
#pragma unroll
        for (int i = 0; i < 2; ++i) {
            int row = waveM + i * 16 + (lane & 15);
            int col = (lane >> 4) * 8;          // K chunks {0..7}/{8..15} + 16
            uint4* p = (uint4*)&aF[i];
            p[0] = *(const uint4*)&As[row][col];
            p[1] = *(const uint4*)&As[row][col + 16];
        }
#pragma unroll
        for (int j = 0; j < NSUB; ++j) {
            int n   = waveN + j * 16 + (lane & 15);
            int col = (lane >> 4) * 8;          // K chunks {0..7}/{8..15} + 16
            uint4* p = (uint4*)&bF[j];
            p[0] = *(const uint4*)&Bs[n][col];
            p[1] = *(const uint4*)&Bs[n][col + 16];
        }
#pragma unroll
        for (int i = 0; i < 2; ++i)
#pragma unroll
            for (int j = 0; j < NSUB; ++j)
                acc[i][j] = __builtin_amdgcn_wmma_f32_16x16x32_bf16(
                    false, aF[i], false, bF[j], (short)0, acc[i][j], false, false);
        __syncthreads();
    }

    // ---- epilogue ----
#pragma unroll
    for (int i = 0; i < 2; ++i) {
#pragma unroll
        for (int j = 0; j < NSUB; ++j) {
            int n = n0 + waveN + j * 16 + (lane & 15);
            float bv = bias ? bias[n] : 0.0f;
#pragma unroll
            for (int r = 0; r < 8; ++r) {
                int m = m0 + waveM + i * 16 + (lane >> 4) * 8 + r;
                float v = acc[i][j][r] * alpha + bv;
                if (relu) v = fmaxf(v, 0.0f);
                Cb[(long long)m * ldc + n] = v;
            }
        }
    }
}

// ---------------- elementwise kernels ----------------

__global__ void k_f32_to_bf16(const float* __restrict__ in,
                              __hip_bfloat16* __restrict__ out, long long n)
{
    long long i = (long long)blockIdx.x * blockDim.x + threadIdx.x;
    if (i < n) out[i] = __float2bfloat16(in[i]);
}

// out[b,s,d] = emb[tok[b,s], d] + PE(s,d)
__global__ void k_embed(const int* __restrict__ tok, const float* __restrict__ emb,
                        float* __restrict__ out)
{
    long long i = (long long)blockIdx.x * 256 + threadIdx.x;     // MS_*D_ total
    int d = (int)(i % D_);
    long long bs = i / D_;
    int s = (int)(bs % S_);
    int t = tok[bs];
    int half = d >> 1;
    float div = __expf(-(float)(2 * half) * (9.210340371976184f / (float)D_));
    float ang = (float)s * div;
    float pe = (d & 1) ? __cosf(ang) : __sinf(ang);
    out[i] = emb[(long long)t * D_ + d] + pe;
}

// legacy softmax over heads (axis=1): per (q,k), across H strided planes; one b per launch
__global__ void k_softmax_heads(const float* __restrict__ sc,
                                __hip_bfloat16* __restrict__ pr)
{
    long long p = (long long)blockIdx.x * 256 + threadIdx.x;     // S_*S_ total
    float v[H_];
    float mx = -1e30f;
#pragma unroll
    for (int h = 0; h < H_; ++h) {
        v[h] = sc[(long long)h * S_ * S_ + p];
        mx = fmaxf(mx, v[h]);
    }
    float sum = 0.0f;
#pragma unroll
    for (int h = 0; h < H_; ++h) { v[h] = __expf(v[h] - mx); sum += v[h]; }
    float inv = 1.0f / sum;
#pragma unroll
    for (int h = 0; h < H_; ++h)
        pr[(long long)h * S_ * S_ + p] = __float2bfloat16(v[h] * inv);
}

// masked-attn branch == causal cumulative mean of V per (b,d)
__global__ void k_cumavg(const float* __restrict__ in, float* __restrict__ out)
{
    int t = blockIdx.x * 256 + threadIdx.x;                      // B_*D_ total
    int b = t / D_, d = t % D_;
    const float* p = in  + (long long)b * S_ * D_ + d;
    float*       o = out + (long long)b * S_ * D_ + d;
    float acc = 0.0f;
    for (int s = 0; s < S_; ++s) {
        acc += p[(long long)s * D_];
        o[(long long)s * D_] = acc / (float)(s + 1);
    }
}

// out = LN(a + r) * g + beta, optional relu; one block per row of 512
__global__ __launch_bounds__(256) void k_ln(
    const float* __restrict__ a, const float* __restrict__ r,
    const float* __restrict__ g, const float* __restrict__ be,
    float* __restrict__ out, int relu)
{
    __shared__ float red[256];
    long long row = blockIdx.x;
    int tid = threadIdx.x;
    const float* pa = a + row * D_;
    const float* pr = r + row * D_;
    float x0 = pa[tid] + pr[tid];
    float x1 = pa[tid + 256] + pr[tid + 256];
    red[tid] = x0 + x1;
    __syncthreads();
    for (int s = 128; s > 0; s >>= 1) { if (tid < s) red[tid] += red[tid + s]; __syncthreads(); }
    float mean = red[0] * (1.0f / (float)D_);
    __syncthreads();
    float d0 = x0 - mean, d1 = x1 - mean;
    red[tid] = d0 * d0 + d1 * d1;
    __syncthreads();
    for (int s = 128; s > 0; s >>= 1) { if (tid < s) red[tid] += red[tid + s]; __syncthreads(); }
    float inv = rsqrtf(red[0] * (1.0f / (float)D_) + 1e-5f);
    float y0 = d0 * inv * g[tid]       + be[tid];
    float y1 = d1 * inv * g[tid + 256] + be[tid + 256];
    if (relu) { y0 = fmaxf(y0, 0.0f); y1 = fmaxf(y1, 0.0f); }
    out[row * D_ + tid]       = y0;
    out[row * D_ + tid + 256] = y1;
}

// ---------------- host-side orchestration ----------------

static inline void conv_bf16(hipStream_t s, const float* in, __hip_bfloat16* out, long long n)
{
    k_f32_to_bf16<<<(unsigned)((n + 255) / 256), 256, 0, s>>>(in, out, n);
}

static inline void gemm(hipStream_t s,
                        const __hip_bfloat16* A, long long lda, long long sA,
                        const __hip_bfloat16* Bm, long long ldb, long long sB,
                        float* C, long long ldc, long long sC,
                        int M, int N, int K, int batch,
                        const float* bias, float alpha, int relu, int opB)
{
    if (N % 128 == 0) {
        dim3 g((unsigned)(N / 128), (unsigned)(M / 128), (unsigned)batch);
        k_gemm_bf16<4><<<g, 256, 0, s>>>((const u16_t*)A, lda, sA, (const u16_t*)Bm, ldb, sB,
                                         C, ldc, sC, M, N, K, bias, alpha, relu, opB);
    } else {
        dim3 g((unsigned)(N / 64), (unsigned)(M / 128), (unsigned)batch);
        k_gemm_bf16<2><<<g, 256, 0, s>>>((const u16_t*)A, lda, sA, (const u16_t*)Bm, ldb, sB,
                                         C, ldc, sC, M, N, K, bias, alpha, relu, opB);
    }
}

extern "C" void kernel_launch(void* const* d_in, const int* in_sizes, int n_in,
                              void* d_out, int out_size, void* d_ws, size_t ws_size,
                              hipStream_t stream)
{
    (void)in_sizes; (void)n_in; (void)out_size; (void)ws_size;
    auto P = [&](int i) { return (const float*)d_in[i]; };
    const int* toks = (const int*)d_in[0];
    const int* tgts = (const int*)d_in[1];

    char* ws = (char*)d_ws;
    auto F  = [&](size_t mib) { return (float*)(ws + mib * 1024ull * 1024ull); };
    auto BF = [&](size_t mib) { return (__hip_bfloat16*)(ws + mib * 1024ull * 1024ull); };

    float *X  = F(0),  *DEC = F(8),  *TA = F(16), *TB = F(24),
          *TC = F(32), *TD  = F(40), *TF = F(48), *SC = F(80);     // SC: [H,S,S] f32, 32 MiB
    __hip_bfloat16 *XB = BF(112), *QB = BF(116), *KB = BF(120), *VB = BF(124),
                   *DB = BF(128), *EB = BF(132), *FB = BF(136),    // FB: MS*FF bf16
                   *PB = BF(152),                                  // probs [H,S,S] bf16
                   *WB = BF(168);                                  // weight staging, 32 MiB

    const long long MD  = (long long)MS_ * D_;
    const long long SS  = (long long)S_ * S_;
    const long long SD  = (long long)S_ * D_;

    // embeddings + positional encoding
    k_embed<<<(unsigned)(MD / 256), 256, 0, stream>>>(toks, P(2), X);
    k_embed<<<(unsigned)(MD / 256), 256, 0, stream>>>(tgts, P(3), DEC);

    // ---------------- encoder ----------------
    for (int l = 0; l < L_; ++l) {
        int w = 4 + l * 16;  // attn q.w
        conv_bf16(stream, X, XB, MD);
        // Q,K,V projections
        conv_bf16(stream, P(w + 0), WB, (long long)D_ * D_);
        gemm(stream, XB, D_, 0, WB, D_, 0, TA, D_, 0, MS_, D_, D_, 1, P(w + 1), 1.f, 0, 0);
        conv_bf16(stream, P(w + 2), WB, (long long)D_ * D_);
        gemm(stream, XB, D_, 0, WB, D_, 0, TB, D_, 0, MS_, D_, D_, 1, P(w + 3), 1.f, 0, 0);
        conv_bf16(stream, P(w + 4), WB, (long long)D_ * D_);
        gemm(stream, XB, D_, 0, WB, D_, 0, TC, D_, 0, MS_, D_, D_, 1, P(w + 5), 1.f, 0, 0);
        conv_bf16(stream, TA, QB, MD);
        conv_bf16(stream, TB, KB, MD);
        conv_bf16(stream, TC, VB, MD);
        // attention per batch b (softmax over heads needs full [H,S,S])
        for (int b = 0; b < B_; ++b) {
            gemm(stream, QB + b * SD, D_, DH_, KB + b * SD, D_, DH_,
                 SC, S_, SS, S_, S_, DH_, H_, nullptr, 0.125f, 0, 0);
            k_softmax_heads<<<(unsigned)(SS / 256), 256, 0, stream>>>(SC, PB);
            gemm(stream, PB, S_, SS, VB + b * SD, D_, DH_,
                 TD + b * SD, D_, DH_, S_, DH_, S_, H_, nullptr, 1.f, 0, 1);
        }
        // O projection + LN1
        conv_bf16(stream, TD, DB, MD);
        conv_bf16(stream, P(w + 6), WB, (long long)D_ * D_);
        gemm(stream, DB, D_, 0, WB, D_, 0, TA, D_, 0, MS_, D_, D_, 1, P(w + 7), 1.f, 0, 0);
        k_ln<<<MS_, 256, 0, stream>>>(X, TA, P(w + 8), P(w + 9), TB, 0);
        // FF
        conv_bf16(stream, TB, DB, MD);
        conv_bf16(stream, P(w + 10), WB, (long long)FF_ * D_);
        gemm(stream, DB, D_, 0, WB, D_, 0, TF, FF_, 0, MS_, FF_, D_, 1, P(w + 11), 1.f, 1, 0);
        conv_bf16(stream, TF, FB, (long long)MS_ * FF_);
        conv_bf16(stream, P(w + 12), WB, (long long)D_ * FF_);
        gemm(stream, FB, FF_, 0, WB, FF_, 0, TA, D_, 0, MS_, D_, FF_, 1, P(w + 13), 1.f, 0, 0);
        k_ln<<<MS_, 256, 0, stream>>>(TB, TA, P(w + 14), P(w + 15), X, 1);  // relu(LN)
    }

    // ---------------- decoder ----------------
    conv_bf16(stream, X, XB, MD);   // encoder output (fixed across dec layers)
    for (int l = 0; l < L_; ++l) {
        int w = 36 + l * 24;
        conv_bf16(stream, DEC, EB, MD);                       // prev decoder stream
        // attn1 (masked): v-proj on X, causal cumulative mean, o-proj (Q/K unused)
        conv_bf16(stream, P(w + 4), WB, (long long)D_ * D_);
        gemm(stream, XB, D_, 0, WB, D_, 0, TA, D_, 0, MS_, D_, D_, 1, P(w + 5), 1.f, 0, 0);
        k_cumavg<<<(B_ * D_) / 256, 256, 0, stream>>>(TA, TC);
        conv_bf16(stream, TC, QB, MD);
        conv_bf16(stream, P(w + 6), WB, (long long)D_ * D_);
        gemm(stream, QB, D_, 0, WB, D_, 0, TD, D_, 0, MS_, D_, D_, 1, P(w + 7), 1.f, 0, 0);
        k_ln<<<MS_, 256, 0, stream>>>(X, TD, P(w + 8), P(w + 9), TA, 0);   // dq -> TA
        // attn2 (cross): Q from dq, K/V from prev
        conv_bf16(stream, TA, DB, MD);
        conv_bf16(stream, P(w + 10), WB, (long long)D_ * D_);
        gemm(stream, DB, D_, 0, WB, D_, 0, TB, D_, 0, MS_, D_, D_, 1, P(w + 11), 1.f, 0, 0);
        conv_bf16(stream, P(w + 12), WB, (long long)D_ * D_);
        gemm(stream, EB, D_, 0, WB, D_, 0, TC, D_, 0, MS_, D_, D_, 1, P(w + 13), 1.f, 0, 0);
        conv_bf16(stream, P(w + 14), WB, (long long)D_ * D_);
        gemm(stream, EB, D_, 0, WB, D_, 0, TD, D_, 0, MS_, D_, D_, 1, P(w + 15), 1.f, 0, 0);
        conv_bf16(stream, TB, QB, MD);
        conv_bf16(stream, TC, KB, MD);
        conv_bf16(stream, TD, VB, MD);
        for (int b = 0; b < B_; ++b) {
            gemm(stream, QB + b * SD, D_, DH_, KB + b * SD, D_, DH_,
                 SC, S_, SS, S_, S_, DH_, H_, nullptr, 0.125f, 0, 0);
            k_softmax_heads<<<(unsigned)(SS / 256), 256, 0, stream>>>(SC, PB);
            gemm(stream, PB, S_, SS, VB + b * SD, D_, DH_,
                 TB + b * SD, D_, DH_, S_, DH_, S_, H_, nullptr, 1.f, 0, 1);
        }
        conv_bf16(stream, TB, DB, MD);
        conv_bf16(stream, P(w + 16), WB, (long long)D_ * D_);
        gemm(stream, DB, D_, 0, WB, D_, 0, TC, D_, 0, MS_, D_, D_, 1, P(w + 17), 1.f, 0, 0);
        k_ln<<<MS_, 256, 0, stream>>>(TA, TC, P(w + 18), P(w + 19), TD, 0); // dv -> TD
        // FF (D->D here) with ln2 reused + relu
        conv_bf16(stream, TD, DB, MD);
        conv_bf16(stream, P(w + 20), WB, (long long)D_ * D_);
        gemm(stream, DB, D_, 0, WB, D_, 0, TB, D_, 0, MS_, D_, D_, 1, P(w + 21), 1.f, 1, 0);
        conv_bf16(stream, TB, EB, MD);
        conv_bf16(stream, P(w + 22), WB, (long long)D_ * D_);
        gemm(stream, EB, D_, 0, WB, D_, 0, TC, D_, 0, MS_, D_, D_, 1, P(w + 23), 1.f, 0, 0);
        k_ln<<<MS_, 256, 0, stream>>>(TD, TC, P(w + 18), P(w + 19), DEC, 1);
    }

    // ---------------- output projection ----------------
    conv_bf16(stream, DEC, DB, MD);
    conv_bf16(stream, P(84), WB, (long long)VOC_ * D_);
    gemm(stream, DB, D_, 0, WB, D_, 0, (float*)d_out, VOC_, 0,
         MS_, VOC_, D_, 1, P(85), 1.f, 0, 0);
}
